// CrossAttentionBlock_3908420239927
// MI455X (gfx1250) — compile-verified, matrix operand order checked by reading, not run on hardware
//
#include <hip/hip_runtime.h>
#include <hip/hip_bf16.h>

// ---------------------------------------------------------------------------
// CrossAttentionBlock for MI455X (gfx1250, wave32, WMMA f16 -> f32 accum)
//   NUM_HEAD=8, HIDDEN=128, INPUT_DIM=256, N1=N2=1024, BATCH=16, HD=1024
// Pipeline:
//   1) cvt:    f32 -> f16 for x1, x2, wq, wk, wv, proj weights
//   2) linear: Q/K/V projections, 64x128 block tile, 4 WMMA/wave/K-step,
//              double-buffered GLOBAL_LOAD_ASYNC_TO_LDS_B128 staging
//   3) attn:   flash attention, software-pipelined K/V staging, 16 WMMA/chunk
//   4) linear: out = attn @ proj^T + b (f32 out)
// ---------------------------------------------------------------------------

typedef __attribute__((ext_vector_type(16))) _Float16 v16h;
typedef __attribute__((ext_vector_type(8)))  _Float16 v8h;
typedef __attribute__((ext_vector_type(8)))  float    v8f;
typedef __attribute__((ext_vector_type(4)))  float    v4f;
typedef __attribute__((ext_vector_type(4)))  int      v4i;

#define HD    1024  // NUM_HEAD * HIDDEN
#define DIMI  256   // INPUT_DIM
#define NHEAD 8
#define DHEAD 128
#define SEQ   1024
#define MROWS 16384 // BATCH * SEQ
#define NKB   (SEQ / 32)

// ---- CDNA5 async global->LDS path (guarded; falls back to load+ds_store) ---
#if defined(__has_builtin)
#  if __has_builtin(__builtin_amdgcn_global_load_async_to_lds_b128) && \
      __has_builtin(__builtin_amdgcn_s_wait_asynccnt)
#    define USE_ASYNC_LDS 1
#  endif
#endif
#ifndef USE_ASYNC_LDS
#  define USE_ASYNC_LDS 0
#endif

#if USE_ASYNC_LDS
#  define ASYNC_WAIT(n) __builtin_amdgcn_s_wait_asynccnt(n)
typedef v4i __attribute__((address_space(1))) gv4i;  // global int4
typedef v4i __attribute__((address_space(3))) lv4i;  // LDS int4
#else
#  define ASYNC_WAIT(n)
#endif

// copy 16 bytes (8 halves) global -> LDS
__device__ __forceinline__ void copy16_to_lds(const _Float16* g, _Float16* s) {
#if USE_ASYNC_LDS
  __builtin_amdgcn_global_load_async_to_lds_b128((gv4i*)g, (lv4i*)s, 0, 0);
#else
  *(v8h*)s = *(const v8h*)g;
#endif
}

__device__ __forceinline__ v16h cat16(v8h lo, v8h hi) {
  return __builtin_shufflevector(lo, hi, 0,1,2,3,4,5,6,7,8,9,10,11,12,13,14,15);
}

__device__ __forceinline__ v8f wmma_f16(v16h a, v16h b, v8f c) {
  // (neg_a, A, neg_b, B, c_mod, C, reuse_a, reuse_b)
  return __builtin_amdgcn_wmma_f32_16x16x32_f16(false, a, false, b, (short)0, c,
                                                false, false);
}

// ---------------------------------------------------------------------------
// f32 -> f16 conversion, 4 elements / thread
// ---------------------------------------------------------------------------
__global__ void cvt_f32_to_f16(const float* __restrict__ src,
                               _Float16* __restrict__ dst, long n4) {
  long i = (long)blockIdx.x * blockDim.x + threadIdx.x;
  if (i < n4) {
    v4f x = *(const v4f*)(src + i * 4);
    _Float16* d = dst + i * 4;
    d[0] = (_Float16)x.x; d[1] = (_Float16)x.y;
    d[2] = (_Float16)x.z; d[3] = (_Float16)x.w;
  }
}

// ---------------------------------------------------------------------------
// Y[M,N] = X[M,K] @ W[N,K]^T + bias[N]
// grid = (M/64, N/128), block = 256 (8 waves).
// Block tile 64x128; wave w owns 16-col tile w and all 64 rows (4 C tiles).
// Double-buffered async LDS staging; last K-step peeled so the steady-state
// loop is branch-free: stage(i+1) -> s_wait_asynccnt 3 -> barrier -> 4 WMMA.
// ---------------------------------------------------------------------------
__global__ __launch_bounds__(256) void linear_kernel(
    const _Float16* __restrict__ X, const _Float16* __restrict__ W,
    const float* __restrict__ bias, _Float16* __restrict__ Yh,
    float* __restrict__ Yf, int M, int N, int K) {
  __shared__ _Float16 sX[2][64][32];    //  8 KB
  __shared__ _Float16 sW[2][128][32];   // 16 KB

  const int tid  = threadIdx.x;
  const int wave = tid >> 5;
  const int lane = tid & 31;
  const int hf   = lane >> 4;   // half-group (fragment K striping)
  const int ln   = lane & 15;   // row (A) / col (B) within 16-tile

  const long row0 = (long)blockIdx.x * 64;
  const int  col0 = blockIdx.y * 128;
  const int  colw = col0 + wave * 16;

  // per-thread staging coordinates (all 256 threads participate, no exec mask)
  const int xr = tid >> 2, xk = (tid & 3) * 8;   // sX: 64x32, 1 x b128
  const int wc = tid >> 1, wk = (tid & 1) * 16;  // sW: 128x32, 2 x b128

  auto stage = [&](int buf, int kb) {
    copy16_to_lds(X + (row0 + xr) * (long)K + kb + xk, &sX[buf][xr][xk]);
    const _Float16* wp = W + (long)(col0 + wc) * K + kb + wk;
    copy16_to_lds(wp,     &sW[buf][wc][wk]);
    copy16_to_lds(wp + 8, &sW[buf][wc][wk + 8]);
  };

  v8f acc[4];
  #pragma unroll
  for (int r = 0; r < 4; ++r) acc[r] = (v8f){};

  auto compute = [&](int buf) {
    // B fragment (shared by the 4 row tiles of this wave)
    v16h b = cat16(*(const v8h*)&sW[buf][wave * 16 + ln][hf * 8],
                   *(const v8h*)&sW[buf][wave * 16 + ln][16 + hf * 8]);
    #pragma unroll
    for (int r = 0; r < 4; ++r) {
      v16h a = cat16(*(const v8h*)&sX[buf][r * 16 + ln][hf * 8],
                     *(const v8h*)&sX[buf][r * 16 + ln][16 + hf * 8]);
      acc[r] = wmma_f16(a, b, acc[r]);
    }
  };

  const int nk = K / 32;  // >= 2 for all uses (8 or 32)
  stage(0, 0);
  for (int i = 0; i < nk - 1; ++i) {
    stage((i & 1) ^ 1, (i + 1) * 32);
    ASYNC_WAIT(3);
    __syncthreads();
    compute(i & 1);
    __syncthreads();
  }
  ASYNC_WAIT(0);
  __syncthreads();
  compute((nk - 1) & 1);

  const float bv = bias[colw + ln];
  #pragma unroll
  for (int r = 0; r < 4; ++r)
    #pragma unroll
    for (int v = 0; v < 8; ++v) {       // C layout: row = v + 8*hf, col = ln
      float y = acc[r][v] + bv;
      long  gr = row0 + r * 16 + v + 8 * hf;
      if (Yh) Yh[gr * (long)N + colw + ln] = (_Float16)y;
      else    Yf[gr * (long)N + colw + ln] = y;
    }
}

// ---------------------------------------------------------------------------
// Flash attention. grid = (SEQ/128, BATCH, NHEAD), block = 256 (8 waves).
// Wave w handles 16 Q rows; all waves stream the same 32-key K/V chunks.
// Software pipeline: async K loads + V register loads for chunk i+1 are
// issued before the 16-WMMA compute of chunk i and retired at the top of
// the next iteration (double-buffered sK / sVt).
// ---------------------------------------------------------------------------
__global__ __launch_bounds__(256) void attn_kernel(
    const _Float16* __restrict__ Qh, const _Float16* __restrict__ Kh,
    const _Float16* __restrict__ Vh, const float* __restrict__ Bmat,
    _Float16* __restrict__ Ah) {
  __shared__ _Float16 sK[2][32][DHEAD];   // [key][d]   16 KB
  __shared__ _Float16 sVt[2][DHEAD][32];  // [d][key]   16 KB (transposed)
  __shared__ _Float16 sP[8][16][32];      // per-wave P  8 KB

  const int tid  = threadIdx.x;
  const int wave = tid >> 5;
  const int lane = tid & 31;
  const int hf   = lane >> 4;
  const int ln   = lane & 15;

  const int  b     = blockIdx.y;
  const int  h     = blockIdx.z;
  const int  qrow0 = blockIdx.x * 128 + wave * 16;  // local q row base
  const long grow0 = (long)b * SEQ + qrow0;         // row into Qh/Ah

  // Preload Q A-fragments: 4 chunks of K=32 across d=128
  v16h qa[4];
  {
    const _Float16* qp = Qh + (grow0 + ln) * (long)HD + h * DHEAD;
    #pragma unroll
    for (int f = 0; f < 4; ++f)
      qa[f] = cat16(*(const v8h*)(qp + f * 32 + hf * 8),
                    *(const v8h*)(qp + f * 32 + 16 + hf * 8));
  }

  // staging coordinates: thread covers (key, 16 d-values) of the 32x128 chunk
  const int key = tid >> 3;
  const int dp  = (tid & 7) * 16;
  const _Float16* kbase = Kh + ((long)b * SEQ + key) * (long)HD + h * DHEAD + dp;
  const _Float16* vbase = Vh + ((long)b * SEQ + key) * (long)HD + h * DHEAD + dp;

  v8h vr0, vr1;  // in-flight V registers for the next chunk
  auto issueK = [&](int buf, int kb) {
    const _Float16* kp = kbase + (long)kb * 32 * HD;
    copy16_to_lds(kp,     &sK[buf][key][dp]);     // async on CDNA5
    copy16_to_lds(kp + 8, &sK[buf][key][dp + 8]);
  };
  auto loadV = [&](int kb) {
    const _Float16* vp = vbase + (long)kb * 32 * HD;
    vr0 = *(const v8h*)(vp);
    vr1 = *(const v8h*)(vp + 8);
  };
  auto storeV = [&](int buf) {  // transposed store for P@V B fragments
    #pragma unroll
    for (int j = 0; j < 8; ++j) sVt[buf][dp + j][key]     = vr0[j];
    #pragma unroll
    for (int j = 0; j < 8; ++j) sVt[buf][dp + 8 + j][key] = vr1[j];
  };

  float m[8], lsum[8];
  v8f o[8];
  #pragma unroll
  for (int v = 0; v < 8; ++v) { m[v] = -3.0e38f; lsum[v] = 0.f; }
  #pragma unroll
  for (int t = 0; t < 8; ++t) o[t] = (v8f){};

  const float scale = 0.08838834764831845f;  // 1/sqrt(128)

  issueK(0, 0);
  loadV(0);
  for (int kb = 0; kb < NKB; ++kb) {
    const int cur = kb & 1;
    ASYNC_WAIT(0);       // K chunk kb resident in sK[cur]
    storeV(cur);         // V chunk kb registers -> sVt[cur]
    __syncthreads();
    if (kb + 1 < NKB) {  // chunk kb+1 in flight across the compute below
      issueK(cur ^ 1, kb + 1);
      loadV(kb + 1);
    }

    // Two 16x16 score tiles covering 32 keys: S = Q @ K^T
    v8f s[2];
    #pragma unroll
    for (int t = 0; t < 2; ++t) {
      v8f c = (v8f){};
      #pragma unroll
      for (int f = 0; f < 4; ++f) {
        const _Float16* kr = &sK[cur][t * 16 + ln][f * 32];
        v16h bb = cat16(*(const v8h*)(kr + hf * 8),
                        *(const v8h*)(kr + 16 + hf * 8));
        c = wmma_f16(qa[f], bb, c);
      }
      s[t] = c;
    }

    // scale + bias: element v maps to (row qrow0+v+8*hf, key kb*32+t*16+ln)
    #pragma unroll
    for (int t = 0; t < 2; ++t)
      #pragma unroll
      for (int v = 0; v < 8; ++v)
        s[t][v] = s[t][v] * scale +
                  Bmat[(long)(qrow0 + v + 8 * hf) * SEQ + kb * 32 + t * 16 + ln];

    // online softmax: rows span the 16 lanes of a half-group
    float mt[8], alpha[8], rs[8];
    #pragma unroll
    for (int v = 0; v < 8; ++v) mt[v] = fmaxf(s[0][v], s[1][v]);
    #pragma unroll
    for (int off = 1; off < 16; off <<= 1)
      #pragma unroll
      for (int v = 0; v < 8; ++v)
        mt[v] = fmaxf(mt[v], __shfl_xor(mt[v], off, 16));
    #pragma unroll
    for (int v = 0; v < 8; ++v) {
      float mn = fmaxf(m[v], mt[v]);
      alpha[v] = __expf(m[v] - mn);
      m[v] = mn;
      float p0 = __expf(s[0][v] - mn);
      float p1 = __expf(s[1][v] - mn);
      s[0][v] = p0; s[1][v] = p1;
      rs[v] = p0 + p1;
    }
    #pragma unroll
    for (int off = 1; off < 16; off <<= 1)
      #pragma unroll
      for (int v = 0; v < 8; ++v) rs[v] += __shfl_xor(rs[v], off, 16);
    #pragma unroll
    for (int v = 0; v < 8; ++v) {
      lsum[v] = lsum[v] * alpha[v] + rs[v];
      #pragma unroll
      for (int t = 0; t < 8; ++t) o[t][v] *= alpha[v];
    }

    // Repack P (C layout) -> A fragment via per-wave LDS transpose buffer
    #pragma unroll
    for (int t = 0; t < 2; ++t)
      #pragma unroll
      for (int v = 0; v < 8; ++v)
        sP[wave][v + 8 * hf][t * 16 + ln] = (_Float16)s[t][v];
    v16h pa = cat16(*(const v8h*)&sP[wave][ln][hf * 8],
                    *(const v8h*)&sP[wave][ln][16 + hf * 8]);

    // O += P @ V   (8 d-tiles of 16)
    #pragma unroll
    for (int t = 0; t < 8; ++t) {
      const _Float16* vr = &sVt[cur][t * 16 + ln][0];
      v16h vb = cat16(*(const v8h*)(vr + hf * 8),
                      *(const v8h*)(vr + 16 + hf * 8));
      o[t] = wmma_f16(pa, vb, o[t]);
    }
    __syncthreads();
  }

  // normalize + store attn output (f16) into [MROWS, HD]
  #pragma unroll
  for (int v = 0; v < 8; ++v) {
    float inv = 1.0f / lsum[v];
    long r = (grow0 + v + 8 * hf) * (long)HD + h * DHEAD;
    #pragma unroll
    for (int t = 0; t < 8; ++t)
      Ah[r + t * 16 + ln] = (_Float16)(o[t][v] * inv);
  }
}

// ---------------------------------------------------------------------------
extern "C" void kernel_launch(void* const* d_in, const int* in_sizes, int n_in,
                              void* d_out, int out_size, void* d_ws,
                              size_t ws_size, hipStream_t stream) {
  const float* x1   = (const float*)d_in[0];
  const float* x2   = (const float*)d_in[1];
  const float* Bm   = (const float*)d_in[2];
  const float* wq_w = (const float*)d_in[3];
  const float* wq_b = (const float*)d_in[4];
  const float* wk_w = (const float*)d_in[5];
  const float* wk_b = (const float*)d_in[6];
  const float* wv_w = (const float*)d_in[7];
  const float* wv_b = (const float*)d_in[8];
  const float* pj_w = (const float*)d_in[9];
  const float* pj_b = (const float*)d_in[10];
  float* out = (float*)d_out;

  const long NX = (long)MROWS * DIMI;  // 4,194,304
  const long NW = (long)HD * DIMI;     // 262,144
  const long NP = (long)DHEAD * HD;    // 131,072
  const long NQ = (long)MROWS * HD;    // 16,777,216

  char* ws = (char*)d_ws;
  size_t off = 0;
  auto alloc = [&](long nelem) {
    void* p = ws + off;
    off += ((size_t)nelem * 2 + 255) & ~(size_t)255;
    return (_Float16*)p;
  };
  _Float16* x1h = alloc(NX);
  _Float16* x2h = alloc(NX);
  _Float16* wqh = alloc(NW);
  _Float16* wkh = alloc(NW);
  _Float16* wvh = alloc(NW);
  _Float16* pjh = alloc(NP);
  _Float16* Qh  = alloc(NQ);
  _Float16* Kh  = alloc(NQ);
  _Float16* Vh  = alloc(NQ);
  _Float16* Ah  = alloc(NQ);
  (void)ws_size; (void)in_sizes; (void)n_in; (void)out_size;

  auto cvt = [&](const float* s, _Float16* d, long n) {
    long n4 = n / 4;
    cvt_f32_to_f16<<<dim3((n4 + 255) / 256), dim3(256), 0, stream>>>(s, d, n4);
  };
  cvt(x1,   x1h, NX);
  cvt(x2,   x2h, NX);
  cvt(wq_w, wqh, NW);
  cvt(wk_w, wkh, NW);
  cvt(wv_w, wvh, NW);
  cvt(pj_w, pjh, NP);

  // QKV projections: [16384,256] x [256,1024] -> f16
  dim3 gq(MROWS / 64, HD / 128);
  linear_kernel<<<gq, 256, 0, stream>>>(x1h, wqh, wq_b, Qh, nullptr,
                                        MROWS, HD, DIMI);
  linear_kernel<<<gq, 256, 0, stream>>>(x2h, wkh, wk_b, Kh, nullptr,
                                        MROWS, HD, DIMI);
  linear_kernel<<<gq, 256, 0, stream>>>(x2h, wvh, wv_b, Vh, nullptr,
                                        MROWS, HD, DIMI);

  // fused attention
  attn_kernel<<<dim3(SEQ / 128, 16, NHEAD), 256, 0, stream>>>(Qh, Kh, Vh, Bm, Ah);

  // output projection: [16384,1024] x [1024,128] -> f32 out
  linear_kernel<<<dim3(MROWS / 64, 1), 256, 0, stream>>>(Ah, pjh, pj_b, nullptr,
                                                         out, MROWS, DHEAD, HD);
}
